// SelfAttention_7404523618858
// MI455X (gfx1250) — compile-verified, matrix operand order checked by reading, not run on hardware
//
#include <hip/hip_runtime.h>

typedef __bf16 bf16;
typedef __attribute__((ext_vector_type(16))) __bf16 v16bf;
typedef __attribute__((ext_vector_type(8)))  __bf16 v8bf;
typedef __attribute__((ext_vector_type(4)))  __bf16 v4bf;
typedef __attribute__((ext_vector_type(8)))  float  v8f;
typedef __attribute__((ext_vector_type(4)))  float  v4f;

#define BB   2
#define SS   2048
#define DD   1024
#define HH   16
#define DH   64
#define MM   (BB * SS)          // 4096 rows
#define E3D  (3 * DD)           // 3072
// 1/sqrt(Dh) * log2(e), pre-folded into Q at projection time
#define QSCALE 0.18033688011112042f

#define WMMA_BF16(A, B, C) __builtin_amdgcn_wmma_f32_16x16x32_bf16( \
    false, (A), false, (B), (short)0, (C), false, false)

// Operand load: `base_k` is wave-uniform, `laneoff` a loop-invariant 32-bit
// per-lane element offset -> backend folds row strides into load immediates.
// Layout per ISA 7.12.2 (16-bit A/B operand, one 16x32 K-slice):
//   lanes 0-15 : K = {0..7, 16..23};  lanes 16-31: K = {8..15, 24..31}
__device__ __forceinline__ v16bf load_op16(const bf16* base_k, int laneoff) {
  const bf16* rp = base_k + laneoff;
  v8bf lo = *(const v8bf*)(rp);
  v8bf hi = *(const v8bf*)(rp + 16);
  v16bf a;
#pragma unroll
  for (int i = 0; i < 8; ++i) { a[i] = lo[i]; a[i + 8] = hi[i]; }
  return a;
}

// One operand set (2 A-halves + 4 B-subtiles) for a 32Mx64N tile at K=kk.
__device__ __forceinline__ void load_set(const bf16* A_k, const bf16* W_k,
                                         int offA0, int offA1, const int* offB,
                                         v16bf& a0, v16bf& a1, v16bf* bm) {
  a0 = load_op16(A_k, offA0);
  a1 = load_op16(A_k, offA1);
#pragma unroll
  for (int nn = 0; nn < 4; ++nn) bm[nn] = load_op16(W_k, offB[nn]);
}

__global__ void __launch_bounds__(256) cvt_f32_bf16(const float* __restrict__ in,
                                                    bf16* __restrict__ out, int n4) {
  int i = blockIdx.x * 256 + threadIdx.x;
  if (i < n4) {
    v4f f = *(const v4f*)(in + (size_t)i * 4);
    v4bf o;
#pragma unroll
    for (int j = 0; j < 4; ++j) o[j] = (bf16)f[j];
    *(v4bf*)(out + (size_t)i * 4) = o;
  }
}

// qkv = x @ w_in^T + b_in ; scatter into Q[b,h,s,dh] (pre-scaled by QSCALE),
// K[b,h,s,dh], V^T[b,h,dh,s]. 32(M) x 64(N) tile per wave, pipelined k-loop.
__global__ void __launch_bounds__(128) qkv_gemm(const bf16* __restrict__ xb,
                                                const bf16* __restrict__ wb,
                                                const float* __restrict__ b_in,
                                                bf16* __restrict__ qb,
                                                bf16* __restrict__ kb,
                                                bf16* __restrict__ vtb) {
  const int lane = threadIdx.x & 31;
  const int wid  = __builtin_amdgcn_readfirstlane(blockIdx.x * 4 + (threadIdx.x >> 5));
  const int NG   = E3D / 64;                 // 48 groups of 64 output features
  const int mt   = wid / NG, ng = wid % NG;
  const int m0   = mt * 32, n0 = ng * 64;
  const int col  = lane & 15;
  const int adj  = (lane & 16) ? 8 : 0;
  const int rhi  = (lane & 16) ? 8 : 0;

  const int offA0 = (m0 + col) * DD + adj;
  const int offA1 = (m0 + 16 + col) * DD + adj;
  int offB[4];
#pragma unroll
  for (int nn = 0; nn < 4; ++nn) offB[nn] = (n0 + nn * 16 + col) * DD + adj;

  v8f acc0[4] = {}, acc1[4] = {};
  v16bf a0c, a1c, bmc[4], a0n, a1n, bmn[4];
  load_set(xb, wb, offA0, offA1, offB, a0c, a1c, bmc);
  for (int kk = 0; kk < DD; kk += 64) {
    load_set(xb + kk + 32, wb + kk + 32, offA0, offA1, offB, a0n, a1n, bmn);
#pragma unroll
    for (int nn = 0; nn < 4; ++nn) acc0[nn] = WMMA_BF16(a0c, bmc[nn], acc0[nn]);
#pragma unroll
    for (int nn = 0; nn < 4; ++nn) acc1[nn] = WMMA_BF16(a1c, bmc[nn], acc1[nn]);
    const int k2 = (kk + 64 < DD) ? kk + 64 : DD - 32;   // clamped dead tail
    load_set(xb + k2, wb + k2, offA0, offA1, offB, a0c, a1c, bmc);
#pragma unroll
    for (int nn = 0; nn < 4; ++nn) acc0[nn] = WMMA_BF16(a0n, bmn[nn], acc0[nn]);
#pragma unroll
    for (int nn = 0; nn < 4; ++nn) acc1[nn] = WMMA_BF16(a1n, bmn[nn], acc1[nn]);
  }

  // Uniform routing: a 64-wide N-group lies in exactly one of {q,k,v} and one
  // head; a 32-row M-tile lies in one batch. All routing values are scalar.
  const int sec  = n0 >> 10;                 // 0=q 1=k 2=v
  const int h    = (n0 & 1023) >> 6;
  const int bidx = m0 >> 11;
  const int s0   = m0 & 2047;
  const size_t bhh = (size_t)(bidx * HH + h);
  const float wscale = (sec == 0) ? QSCALE : 1.0f;   // fold softmax scale into Q

  if (sec < 2) {
    bf16* __restrict__ dst = (sec == 0 ? qb : kb) + bhh * SS * DH;
#pragma unroll
    for (int half = 0; half < 2; ++half) {
      const v8f* acc = half ? acc1 : acc0;
#pragma unroll
      for (int nn = 0; nn < 4; ++nn) {
        const int dh = nn * 16 + col;
        const float bias = b_in[n0 + dh];
#pragma unroll
        for (int r = 0; r < 8; ++r) {
          const int s = s0 + half * 16 + r + rhi;
          dst[(size_t)s * DH + dh] = (bf16)((acc[nn][r] + bias) * wscale);
        }
      }
    }
  } else {
    bf16* __restrict__ dst = vtb + bhh * DH * SS;
#pragma unroll
    for (int half = 0; half < 2; ++half) {
      const v8f* acc = half ? acc1 : acc0;
#pragma unroll
      for (int nn = 0; nn < 4; ++nn) {
        const int dh = nn * 16 + col;
        const float bias = b_in[n0 + dh];
#pragma unroll
        for (int r = 0; r < 8; ++r) {
          const int s = s0 + half * 16 + r + rhi;
          dst[(size_t)dh * SS + s] = (bf16)(acc[nn][r] + bias);
        }
      }
    }
  }
}

// Flash-attention: one wave handles one (b, h, 16-query tile).
__global__ void __launch_bounds__(128) attn_fa(const bf16* __restrict__ qb,
                                               const bf16* __restrict__ kb,
                                               const bf16* __restrict__ vtb,
                                               bf16* __restrict__ attnb) {
  __shared__ __align__(16) bf16 plds[4][16 * 32];   // P transpose staging, 1KB/wave
  const int lane = threadIdx.x & 31;
  const int wave = threadIdx.x >> 5;
  const int wid  = __builtin_amdgcn_readfirstlane(blockIdx.x * 4 + wave);
  const int b  = wid >> 11;
  const int h  = (wid >> 7) & 15;
  const int q0 = (wid & 127) * 16;
  const size_t bh = (size_t)(b * HH + h);
  const bf16* qbase = qb  + bh * SS * DH;
  const bf16* kbase = kb  + bh * SS * DH;
  const bf16* vbase = vtb + bh * DH * SS;
  const int col = lane & 15;
  const int adj = (lane & 16) ? 8 : 0;
  const int rhi = (lane & 16) ? 8 : 0;

  const int offQ = col * DH + adj;
  int offK[2], offV[4];
#pragma unroll
  for (int nh = 0; nh < 2; ++nh) offK[nh] = (nh * 16 + col) * DH + adj;
#pragma unroll
  for (int nv = 0; nv < 4; ++nv) offV[nv] = (nv * 16 + col) * SS + adj;

  // Q tile (16 x 64), pre-scaled by QSCALE at projection time
  v16bf qa0 = load_op16(qbase + q0 * DH,      offQ);
  v16bf qa1 = load_op16(qbase + q0 * DH + 32, offQ);

  v8f oacc[4] = {};
  float rmax[8], rsum[8];    // rsum is a PER-LANE partial (reduced once at end)
#pragma unroll
  for (int r = 0; r < 8; ++r) { rmax[r] = -1e30f; rsum[r] = 0.0f; }

  bf16* pl = &plds[wave][0];
  const int nfull = q0 >> 5;   // blocks fully below the diagonal (no masking)

  // prologue: K operands for block 0
  v16bf km[4];
#pragma unroll
  for (int nh = 0; nh < 2; ++nh) {
    km[nh * 2 + 0] = load_op16(kbase,      offK[nh]);
    km[nh * 2 + 1] = load_op16(kbase + 32, offK[nh]);
  }

  auto kblock = [&](int j0, bool masked) {
    // S = Q K^T for 16 q x 32 k
    v8f sacc[2] = {};
#pragma unroll
    for (int nh = 0; nh < 2; ++nh) {
      sacc[nh] = WMMA_BF16(qa0, km[nh * 2 + 0], sacc[nh]);
      sacc[nh] = WMMA_BF16(qa1, km[nh * 2 + 1], sacc[nh]);
    }
    // prefetch next K block (clamped dead tail) — overlaps the softmax VALU
    const int jn = (j0 + 32 < SS) ? j0 + 32 : SS - 32;
    const bf16* kblk = kbase + jn * DH;
#pragma unroll
    for (int nh = 0; nh < 2; ++nh) {
      km[nh * 2 + 0] = load_op16(kblk,      offK[nh]);
      km[nh * 2 + 1] = load_op16(kblk + 32, offK[nh]);
    }
    // online softmax (exp2 domain; scale already folded into Q)
#pragma unroll
    for (int r = 0; r < 8; ++r) {
      float s0 = sacc[0][r];
      float s1 = sacc[1][r];
      if (masked) {
        const int qrow = q0 + r + rhi;
        if (j0 + col      > qrow) s0 = -1e30f;
        if (j0 + 16 + col > qrow) s1 = -1e30f;
      }
      float m = fmaxf(s0, s1);
#pragma unroll
      for (int xm = 1; xm < 16; xm <<= 1) m = fmaxf(m, __shfl_xor(m, xm, 32));
      const float nm   = fmaxf(rmax[r], m);
      const float corr = exp2f(rmax[r] - nm);
      rmax[r] = nm;
      const float p0 = exp2f(s0 - nm), p1 = exp2f(s1 - nm);
      rsum[r] = rsum[r] * corr + (p0 + p1);      // per-lane partial sum
#pragma unroll
      for (int nv = 0; nv < 4; ++nv) oacc[nv][r] *= corr;
      const int ml = r + rhi;                    // C-layout row -> LDS row
      pl[ml * 32 + col]      = (bf16)p0;
      pl[ml * 32 + 16 + col] = (bf16)p1;
    }
    // issue V-tile global loads first so they overlap the LDS wait
    v16bf bv[4];
    const bf16* vblk = vbase + j0;
#pragma unroll
    for (int nv = 0; nv < 4; ++nv)
      bv[nv] = load_op16(vblk, offV[nv]);        // V^T rows
    asm volatile("s_wait_dscnt 0" ::: "memory"); // P stores visible to reads
    // re-read P (16x32) in WMMA A-layout, then O += P V
    const bf16* rp = pl + (lane & 15) * 32 + adj;
    v8bf lo = *(const v8bf*)rp;
    v8bf hi = *(const v8bf*)(rp + 16);
    v16bf pa;
#pragma unroll
    for (int i = 0; i < 8; ++i) { pa[i] = lo[i]; pa[i + 8] = hi[i]; }
#pragma unroll
    for (int nv = 0; nv < 4; ++nv)
      oacc[nv] = WMMA_BF16(pa, bv[nv], oacc[nv]);
  };

  for (int blk = 0; blk < nfull; ++blk) kblock(blk << 5, false);
  kblock(nfull << 5, true);                      // single diagonal block

  // final per-row sum reduction, normalize, store as bf16 [B,S,D]
  bf16* __restrict__ abase = attnb + ((size_t)b * SS + q0) * DD + h * DH;
#pragma unroll
  for (int r = 0; r < 8; ++r) {
    float t = rsum[r];
#pragma unroll
    for (int xm = 1; xm < 16; xm <<= 1) t += __shfl_xor(t, xm, 32);
    const float inv = 1.0f / t;
    const size_t rowoff = (size_t)(r + rhi) * DD;
#pragma unroll
    for (int nv = 0; nv < 4; ++nv)
      abase[rowoff + nv * 16 + col] = (bf16)(oacc[nv][r] * inv);
  }
}

// out = attn @ w_out^T + b_out (fp32 result); 32(M) x 64(N) per wave, pipelined
__global__ void __launch_bounds__(128) out_gemm(const bf16* __restrict__ ab,
                                                const bf16* __restrict__ wb,
                                                const float* __restrict__ b_out,
                                                float* __restrict__ out) {
  const int lane = threadIdx.x & 31;
  const int wid  = __builtin_amdgcn_readfirstlane(blockIdx.x * 4 + (threadIdx.x >> 5));
  const int NG   = DD / 64;                   // 16
  const int mt   = wid / NG, ng = wid % NG;
  const int m0   = mt * 32, n0 = ng * 64;
  const int col  = lane & 15;
  const int adj  = (lane & 16) ? 8 : 0;
  const int rhi  = (lane & 16) ? 8 : 0;

  const int offA0 = (m0 + col) * DD + adj;
  const int offA1 = (m0 + 16 + col) * DD + adj;
  int offB[4];
#pragma unroll
  for (int nn = 0; nn < 4; ++nn) offB[nn] = (n0 + nn * 16 + col) * DD + adj;

  v8f acc0[4] = {}, acc1[4] = {};
  v16bf a0c, a1c, bmc[4], a0n, a1n, bmn[4];
  load_set(ab, wb, offA0, offA1, offB, a0c, a1c, bmc);
  for (int kk = 0; kk < DD; kk += 64) {
    load_set(ab + kk + 32, wb + kk + 32, offA0, offA1, offB, a0n, a1n, bmn);
#pragma unroll
    for (int nn = 0; nn < 4; ++nn) acc0[nn] = WMMA_BF16(a0c, bmc[nn], acc0[nn]);
#pragma unroll
    for (int nn = 0; nn < 4; ++nn) acc1[nn] = WMMA_BF16(a1c, bmc[nn], acc1[nn]);
    const int k2 = (kk + 64 < DD) ? kk + 64 : DD - 32;   // clamped dead tail
    load_set(ab + k2, wb + k2, offA0, offA1, offB, a0c, a1c, bmc);
#pragma unroll
    for (int nn = 0; nn < 4; ++nn) acc0[nn] = WMMA_BF16(a0n, bmn[nn], acc0[nn]);
#pragma unroll
    for (int nn = 0; nn < 4; ++nn) acc1[nn] = WMMA_BF16(a1n, bmn[nn], acc1[nn]);
  }
  float* __restrict__ obase = out + (size_t)m0 * DD + n0;
#pragma unroll
  for (int half = 0; half < 2; ++half) {
    const v8f* acc = half ? acc1 : acc0;
#pragma unroll
    for (int nn = 0; nn < 4; ++nn) {
      const int e = nn * 16 + col;
      const float bias = b_out[n0 + e];
#pragma unroll
      for (int r = 0; r < 8; ++r)
        obase[(size_t)(half * 16 + r + rhi) * DD + e] = acc[nn][r] + bias;
    }
  }
}

extern "C" void kernel_launch(void* const* d_in, const int* in_sizes, int n_in,
                              void* d_out, int out_size, void* d_ws, size_t ws_size,
                              hipStream_t stream) {
  const float* x     = (const float*)d_in[0];
  const float* w_in  = (const float*)d_in[1];
  const float* b_in  = (const float*)d_in[2];
  const float* w_out = (const float*)d_in[3];
  const float* b_out = (const float*)d_in[4];
  float* out = (float*)d_out;

  char* p = (char*)d_ws;
  bf16* xb     = (bf16*)p; p += (size_t)MM * DD * 2;           // 8 MB
  bf16* w_inb  = (bf16*)p; p += (size_t)E3D * DD * 2;          // 6 MB
  bf16* w_outb = (bf16*)p; p += (size_t)DD * DD * 2;           // 2 MB
  bf16* qb     = (bf16*)p; p += (size_t)BB * HH * SS * DH * 2; // 8 MB
  bf16* kb     = (bf16*)p; p += (size_t)BB * HH * SS * DH * 2; // 8 MB
  bf16* vtb    = (bf16*)p; p += (size_t)BB * HH * SS * DH * 2; // 8 MB
  bf16* attnb  = (bf16*)p;                                     // 8 MB  (48 MB total)

  cvt_f32_bf16<<<(MM * DD / 4 + 255) / 256, 256, 0, stream>>>(x, xb, MM * DD / 4);
  cvt_f32_bf16<<<(E3D * DD / 4 + 255) / 256, 256, 0, stream>>>(w_in, w_inb, E3D * DD / 4);
  cvt_f32_bf16<<<(DD * DD / 4 + 255) / 256, 256, 0, stream>>>(w_out, w_outb, DD * DD / 4);

  // QKV: 128 M-tiles(32) x 48 N-groups(64) = 6144 waves, 4 waves/block
  qkv_gemm<<<(128 * 48) / 4, 128, 0, stream>>>(xb, w_inb, b_in, qb, kb, vtb);

  // Attention: 2*16*128 = 4096 q-tile waves, 4 waves/block
  attn_fa<<<4096 / 4, 128, 0, stream>>>(qb, kb, vtb, attnb);

  // Out projection: 128 M-tiles(32) x 16 N-groups(64) = 2048 waves
  out_gemm<<<(128 * 16) / 4, 128, 0, stream>>>(attnb, w_outb, b_out, out);
}
